// FramePrimerDecoder_1279900254349
// MI455X (gfx1250) — compile-verified, matrix operand order checked by reading, not run on hardware
//
#include <hip/hip_runtime.h>
#include <math.h>

// ---------------- problem constants ----------------
static constexpr int Bn    = 4;
static constexpr int CHn   = 4;
static constexpr int BINSn = 1024;
static constexpr int CROPn = 1024;
static constexpr int NBn   = 4;
static constexpr int FFn   = 2048;
static constexpr int HDn   = BINSn / NBn;          // 256
static constexpr long PL   = (long)BINSn * CROPn;  // 1,048,576 plane

// epilogue mode flags
static constexpr int EP_ACC    = 1;  // val += C[idx] (read-modify-write)
static constexpr int EP_BF16   = 2;  // store bf16 to Cbf
static constexpr int EP_RELU2  = 4;  // val = relu(val)^2
static constexpr int EP_TRANSC = 8;  // store C transposed: C[n*ldc + m]

// ---------------- WMMA types ----------------
typedef __attribute__((ext_vector_type(16))) __bf16 bf16x16;
typedef __attribute__((ext_vector_type(8)))  __bf16 bf16x8;
typedef __attribute__((ext_vector_type(8)))  float  f32x8;

__device__ __forceinline__ f32x8 zero8() {
  f32x8 v;
#pragma unroll
  for (int i = 0; i < 8; ++i) v[i] = 0.0f;
  return v;
}

// Fragment K layout for v_wmma_f32_16x16x32_bf16 (ISA 7.12.2, 16-bit 16x32):
// per lane, K indices form two contiguous runs of 8:
//   [hi*8 .. hi*8+7] and [16+hi*8 .. 16+hi*8+7], hi = lane>>4.
// With an LDS row holding 32 contiguous K values, a fragment = two 16B loads.
__device__ __forceinline__ bf16x16 load_frag(const __bf16* row, int lane) {
  int hi = (lane >> 4) * 8;
  bf16x8 lo8 = *reinterpret_cast<const bf16x8*>(row + hi);
  bf16x8 hi8 = *reinterpret_cast<const bf16x8*>(row + 16 + hi);
  bf16x16 f;
#pragma unroll
  for (int e = 0; e < 8; ++e) { f[e] = lo8[e]; f[8 + e] = hi8[e]; }
  return f;
}

// ---------------- bf16 WMMA GEMM, double-buffered LDS ----------------
// C(M,N) = A(M,K) @ B(K,N); A,B bf16 in memory; A is always (M,K) row-major.
// TRANSB=true: B memory is (N,K) row-major (logical B[k][n] = Bm[n*ldb+k]).
// Batched over grid.z = nb*nh;  z -> (b = z/nh, h = z%nh) pointer offsets.
// Epilogue: val = acc; if ACC val += C; val *= alpha; if prevp val += prev;
//           if RELU2 val = relu(val)^2; store fp32 (C) or bf16 (Cbf);
//           TRANSC stores at C[n*ldc + m] (per-lane contiguous m-run of 8).
template <bool TRANSB>
__global__ __launch_bounds__(256) void gemm_bf16_kernel(
    int M, int N, int K,
    const __bf16* __restrict__ A, int lda, long aSB, long aSH,
    const __bf16* __restrict__ Bm, int ldb, long bSB, long bSH,
    float* C, __bf16* Cbf, int mode, float alpha, const float* __restrict__ prevp,
    int ldc, long cSB, long cSH, int nh)
{
  __shared__ __bf16 As[2][128][40];   // m-major, 32 K + 8 pad (16B-aligned rows)
  __shared__ __bf16 Bs[2][128][40];   // n-major, 32 K + 8 pad

  const int z  = blockIdx.z;
  const int bb = z / nh, hh = z % nh;
  const __bf16* Ab = A  + (long)bb * aSB + (long)hh * aSH;
  const __bf16* Bb = Bm + (long)bb * bSB + (long)hh * bSH;

  const int m0 = blockIdx.y * 128;
  const int n0 = blockIdx.x * 128;

  const int tid  = threadIdx.x;
  const int lane = tid & 31;
  const int wave = tid >> 5;
  const int wrow = (wave >> 2) * 64;  // 2 waves tile M
  const int wcol = (wave & 3) * 32;   // 4 waves tile N

  f32x8 acc[4][2];
#pragma unroll
  for (int i = 0; i < 4; ++i)
#pragma unroll
    for (int j = 0; j < 2; ++j) acc[i][j] = zero8();

  bf16x8 arg0, arg1, brg0, brg1;  // staging registers (32B per matrix per thread)

  auto gload = [&](int k0) {
    {
      int r = tid >> 1, cb = (tid & 1) * 16;
      const __bf16* s = Ab + (long)(m0 + r) * lda + k0 + cb;
      arg0 = *reinterpret_cast<const bf16x8*>(s);
      arg1 = *reinterpret_cast<const bf16x8*>(s + 8);
    }
    if (!TRANSB) {
      int k = tid >> 3, nb = (tid & 7) * 16;
      const __bf16* s = Bb + (long)(k0 + k) * ldb + (n0 + nb);
      brg0 = *reinterpret_cast<const bf16x8*>(s);
      brg1 = *reinterpret_cast<const bf16x8*>(s + 8);
    } else {
      int n = tid >> 1, kb = (tid & 1) * 16;
      const __bf16* s = Bb + (long)(n0 + n) * ldb + k0 + kb;
      brg0 = *reinterpret_cast<const bf16x8*>(s);
      brg1 = *reinterpret_cast<const bf16x8*>(s + 8);
    }
  };

  auto sstore = [&](int buf) {
    {
      int r = tid >> 1, cb = (tid & 1) * 16;
      *reinterpret_cast<bf16x8*>(&As[buf][r][cb])     = arg0;
      *reinterpret_cast<bf16x8*>(&As[buf][r][cb + 8]) = arg1;
    }
    if (!TRANSB) {
      int k = tid >> 3, nb = (tid & 7) * 16;
#pragma unroll
      for (int j = 0; j < 8; ++j) {
        Bs[buf][nb + j][k]     = brg0[j];
        Bs[buf][nb + 8 + j][k] = brg1[j];
      }
    } else {
      int n = tid >> 1, kb = (tid & 1) * 16;
      *reinterpret_cast<bf16x8*>(&Bs[buf][n][kb])     = brg0;
      *reinterpret_cast<bf16x8*>(&Bs[buf][n][kb + 8]) = brg1;
    }
  };

  const int nk = K >> 5;
  gload(0);
  sstore(0);
  __syncthreads();

  for (int kt = 0; kt < nk; ++kt) {
    const int cur = kt & 1;
    if (kt + 1 < nk) gload((kt + 1) << 5);       // loads in flight during wmma
    if (kt + 2 < nk) {                           // warm L2 two tiles ahead
      __builtin_prefetch(Ab + (long)(m0 + (tid >> 1)) * lda + (kt + 2) * 32, 0, 1);
      __builtin_prefetch(Bb + (long)(TRANSB ? (n0 + (tid >> 1)) * (long)ldb + (kt + 2) * 32
                                            : ((kt + 2) * 32 + (tid >> 3)) * (long)ldb + n0), 0, 1);
    }

    bf16x16 afrg[4], bfrg[2];
#pragma unroll
    for (int ms = 0; ms < 4; ++ms)
      afrg[ms] = load_frag(&As[cur][wrow + ms * 16 + (lane & 15)][0], lane);
#pragma unroll
    for (int ns = 0; ns < 2; ++ns)
      bfrg[ns] = load_frag(&Bs[cur][wcol + ns * 16 + (lane & 15)][0], lane);
#pragma unroll
    for (int ms = 0; ms < 4; ++ms)
#pragma unroll
      for (int ns = 0; ns < 2; ++ns)
        acc[ms][ns] = __builtin_amdgcn_wmma_f32_16x16x32_bf16(
            false, afrg[ms], false, bfrg[ns], (short)0, acc[ms][ns], false, false);

    if (kt + 1 < nk) sstore(cur ^ 1);            // other buffer: no barrier needed before
    __syncthreads();                             // one barrier per K-step
  }

  // ---- epilogue: C/D layout N = lane&15, M = v + (lane>=16 ? 8 : 0) ----
  const int mofs = (lane >> 4) * 8;
  const long cOff = (long)bb * cSB + (long)hh * cSH;
#pragma unroll
  for (int ms = 0; ms < 4; ++ms)
#pragma unroll
    for (int ns = 0; ns < 2; ++ns) {
      int nn  = n0 + wcol + ns * 16 + (lane & 15);
      int mmb = m0 + wrow + ms * 16 + mofs;
      if (mode & EP_TRANSC) {
        long base = cOff + (long)nn * ldc + mmb;  // per-lane contiguous 8 floats
#pragma unroll
        for (int v = 0; v < 8; ++v) C[base + v] = acc[ms][ns][v] * alpha;
      } else {
#pragma unroll
        for (int v = 0; v < 8; ++v) {
          long idx = cOff + (long)(mmb + v) * ldc + nn;
          float val = acc[ms][ns][v];
          if (mode & EP_ACC)  val += C[idx];
          val *= alpha;
          if (prevp)          val += prevp[idx];
          if (mode & EP_RELU2) { val = fmaxf(val, 0.f); val = val * val; }
          if (mode & EP_BF16) Cbf[idx] = (__bf16)val;
          else                C[idx]   = val;
        }
      }
    }
}

// ---------------- instance-norm stats: mean / rstd per (b, crop) ----------------
__global__ __launch_bounds__(256) void stats_kernel(
    const float* __restrict__ base, long bStride, long chStride, int nch,
    float* __restrict__ mu, float* __restrict__ rs)
{
  int b = blockIdx.y;
  int crop = blockIdx.x * 256 + threadIdx.x;
  const float* p = base + (long)b * bStride + crop;
  float s = 0.f, s2 = 0.f;
  for (int c = 0; c < nch; ++c) {
    const float* pc = p + (long)c * chStride;
    for (int bin = 0; bin < BINSn; ++bin) {
      float v = pc[(long)bin * CROPn];
      s += v; s2 += v * v;
    }
  }
  float inv = 1.0f / (float)(nch * BINSn);
  float m = s * inv;
  float var = fmaxf(s2 * inv - m * m, 0.f);
  mu[b * CROPn + crop] = m;
  rs[b * CROPn + crop] = rsqrtf(var + 1e-5f);
}

// ---------------- inorm + channel mix -> (b, BINS, CROP) fp32 ----------------
__global__ __launch_bounds__(256) void project_kernel(
    const float* __restrict__ base, long bStride, long chStride, int nch,
    const float* __restrict__ mu, const float* __restrict__ rs,
    const float* __restrict__ g, const float* __restrict__ bsh,
    const float* __restrict__ w,
    float* __restrict__ out, long outBStride)
{
  int b = blockIdx.z, bin = blockIdx.y;
  int crop = blockIdx.x * 256 + threadIdx.x;
  float m = mu[b * CROPn + crop], r = rs[b * CROPn + crop];
  float gg = g[crop] * r, bv = bsh[crop];
  const float* p = base + (long)b * bStride + (long)bin * CROPn + crop;
  float acc = 0.f;
  for (int c = 0; c < nch; ++c)
    acc += ((p[(long)c * chStride] - m) * gg + bv) * w[c];
  out[(long)b * outBStride + (long)bin * CROPn + crop] = acc;
}

// ---------------- inorm + channel mix -> transposed (b, CROP, BINS) bf16 ----------------
__global__ __launch_bounds__(256) void premix_kernel(
    const float* __restrict__ base, long bStride, long chStride, int nch,
    const float* __restrict__ mu, const float* __restrict__ rs,
    const float* __restrict__ g, const float* __restrict__ bsh,
    const float* __restrict__ w, const float* __restrict__ bias,
    __bf16* __restrict__ outT)
{
  __shared__ float lds[32][33];
  int b = blockIdx.z;
  int i0 = blockIdx.x * 32;  // crop
  int j0 = blockIdx.y * 32;  // bins
  int tx = threadIdx.x, ty = threadIdx.y;
  int crop = i0 + tx;
  float m = mu[b * CROPn + crop], r = rs[b * CROPn + crop];
  float gg = g[crop] * r, bv = bsh[crop];
  float b0 = bias[0];
#pragma unroll
  for (int k = 0; k < 4; ++k) {
    int binl = ty * 4 + k;
    const float* p = base + (long)b * bStride + (long)(j0 + binl) * CROPn + crop;
    float acc = b0;
    for (int c = 0; c < nch; ++c)
      acc += ((p[(long)c * chStride] - m) * gg + bv) * w[c];
    lds[binl][tx] = acc;
  }
  __syncthreads();
#pragma unroll
  for (int k = 0; k < 4; ++k) {
    int cl = ty * 4 + k;
    outT[((long)b * CROPn + i0 + cl) * BINSn + j0 + tx] = (__bf16)lds[tx][cl];
  }
}

// ---------------- 32x32 tiled transpose: (b,R,C) fp32 -> (b,C,R) bf16 ----------------
// Also used (with gridDim.z==1, strides 0) to pre-transpose weights at bf16 convert time.
__global__ __launch_bounds__(256) void transpose_bf_kernel(
    const float* __restrict__ in, long inBStride,
    __bf16* __restrict__ out, long outBStride, int R, int C)
{
  __shared__ float lds[32][33];
  int b = blockIdx.z;
  int c0 = blockIdx.x * 32, r0 = blockIdx.y * 32;
  int tx = threadIdx.x, ty = threadIdx.y;
#pragma unroll
  for (int k = 0; k < 4; ++k) {
    int rl = ty * 4 + k;
    lds[rl][tx] = in[(long)b * inBStride + (long)(r0 + rl) * C + c0 + tx];
  }
  __syncthreads();
#pragma unroll
  for (int k = 0; k < 4; ++k) {
    int cl = ty * 4 + k;
    out[(long)b * outBStride + (long)(c0 + cl) * R + r0 + tx] = (__bf16)lds[tx][cl];
  }
}

// ---------------- depthwise conv3 along w; fp32 in (B,CROP,BINS) -> bf16 out ----------------
__global__ __launch_bounds__(256) void dwconv_kernel(
    const float* __restrict__ in, const float* __restrict__ kw,
    __bf16* __restrict__ out)
{
  int b = blockIdx.z, w = blockIdx.y;
  int c = blockIdx.x * 256 + threadIdx.x;
  const float* p = in + ((long)b * CROPn + w) * BINSn + c;
  float k0 = kw[c * 3 + 0], k1 = kw[c * 3 + 1], k2 = kw[c * 3 + 2];
  float acc = k1 * p[0];
  if (w > 0)         acc += k0 * p[-(long)BINSn];
  if (w < CROPn - 1) acc += k2 * p[(long)BINSn];
  out[((long)b * CROPn + w) * BINSn + c] = (__bf16)acc;
}

// ---------------- row softmax: (rows, CROP) fp32 -> bf16 probs ----------------
__global__ __launch_bounds__(256) void softmax_kernel(
    const float* __restrict__ qk, __bf16* __restrict__ probs)
{
  __shared__ float red[256];
  long row = blockIdx.x;
  const float* p = qk + row * CROPn;
  __bf16* o = probs + row * CROPn;
  int t = threadIdx.x;
  float mx = -INFINITY;
  float v[4];
#pragma unroll
  for (int u = 0; u < 4; ++u) { v[u] = p[t + u * 256]; mx = fmaxf(mx, v[u]); }
  red[t] = mx; __syncthreads();
  for (int s = 128; s > 0; s >>= 1) { if (t < s) red[t] = fmaxf(red[t], red[t + s]); __syncthreads(); }
  mx = red[0]; __syncthreads();
  float sum = 0.f;
#pragma unroll
  for (int u = 0; u < 4; ++u) { v[u] = __expf(v[u] - mx); sum += v[u]; }
  red[t] = sum; __syncthreads();
  for (int s = 128; s > 0; s >>= 1) { if (t < s) red[t] += red[t + s]; __syncthreads(); }
  float invs = 1.0f / red[0];
#pragma unroll
  for (int u = 0; u < 4; ++u) o[t + u * 256] = (__bf16)(v[u] * invs);
}

// ---------------- host helpers ----------------
static inline void launch_gemm(hipStream_t s, bool transB,
                               int M, int N, int K,
                               const __bf16* A, int lda, long aSB, long aSH,
                               const __bf16* Bm, int ldb, long bSB, long bSH,
                               float* C, __bf16* Cbf, int mode, float alpha, const float* prevp,
                               int ldc, long cSB, long cSH, int nb, int nh)
{
  dim3 g(N / 128, M / 128, nb * nh), blk(256);
  if (transB)
    gemm_bf16_kernel<true><<<g, blk, 0, s>>>(M, N, K, A, lda, aSB, aSH,
        Bm, ldb, bSB, bSH, C, Cbf, mode, alpha, prevp, ldc, cSB, cSH, nh);
  else
    gemm_bf16_kernel<false><<<g, blk, 0, s>>>(M, N, K, A, lda, aSB, aSH,
        Bm, ldb, bSB, bSH, C, Cbf, mode, alpha, prevp, ldc, cSB, cSH, nh);
}

// All projection weights are pre-transposed to (N,K) bf16 -> TRANSB vector staging.
static void run_attention(hipStream_t s,
    const __bf16* xin, const __bf16* memin,
    const __bf16* wqT, const float* cq, const __bf16* wkT, const float* ck,
    const __bf16* wvT, const float* cv, const __bf16* woT, const __bf16* relT,
    const float* prev, float* qk_out, float* xc_ch, long xcBStride,
    float* tmp, __bf16* qb, __bf16* kb, __bf16* vb,
    __bf16* probs, __bf16* merged)
{
  dim3 cgrid(BINSn / 256, CROPn, Bn), cblk(256);
  // Q/K/V projections (bf16 GEMM -> fp32) + depthwise conv3 (-> bf16)
  launch_gemm(s, true, CROPn, BINSn, BINSn, xin, BINSn, PL, 0, wqT, BINSn, 0, 0,
              tmp, nullptr, 0, 1.0f, nullptr, BINSn, PL, 0, Bn, 1);
  dwconv_kernel<<<cgrid, cblk, 0, s>>>(tmp, cq, qb);
  launch_gemm(s, true, CROPn, BINSn, BINSn, memin, BINSn, PL, 0, wkT, BINSn, 0, 0,
              tmp, nullptr, 0, 1.0f, nullptr, BINSn, PL, 0, Bn, 1);
  dwconv_kernel<<<cgrid, cblk, 0, s>>>(tmp, ck, kb);
  launch_gemm(s, true, CROPn, BINSn, BINSn, memin, BINSn, PL, 0, wvT, BINSn, 0, 0,
              tmp, nullptr, 0, 1.0f, nullptr, BINSn, PL, 0, Bn, 1);
  dwconv_kernel<<<cgrid, cblk, 0, s>>>(tmp, cv, vb);
  // S1 = Q K^T -> qk region (fp32, per b,h)
  launch_gemm(s, true, CROPn, CROPn, HDn, qb, BINSn, PL, HDn, kb, BINSn, PL, HDn,
              qk_out, nullptr, 0, 1.0f, nullptr, CROPn, (long)NBn * PL, PL, Bn, NBn);
  // S2T[i,j] = sum_d relT[i,d] q[j,d]; accumulate: qk = (S1 + S2T)/32 + prev
  launch_gemm(s, true, CROPn, CROPn, HDn, relT, HDn, 0, 0, qb, BINSn, PL, HDn,
              qk_out, nullptr, EP_ACC, 0.03125f, prev, CROPn, (long)NBn * PL, PL, Bn, NBn);
  // probs = softmax(qk) (bf16)
  softmax_kernel<<<dim3(Bn * NBn * CROPn), dim3(256), 0, s>>>(qk_out, probs);
  // O = probs @ V -> head-merged (b, w, c) bf16 (only NN-staging GEMM left)
  launch_gemm(s, false, CROPn, HDn, CROPn, probs, CROPn, (long)NBn * PL, PL,
              vb, BINSn, PL, HDn, nullptr, merged, EP_BF16, 1.0f, nullptr, BINSn, PL, HDn, Bn, NBn);
  // xc channel plane = (O @ wo)^T, stored directly transposed (bins, crop)
  launch_gemm(s, true, CROPn, BINSn, BINSn, merged, BINSn, PL, 0, woT, BINSn, 0, 0,
              xc_ch, nullptr, EP_TRANSC, 1.0f, nullptr, CROPn, xcBStride, 0, Bn, 1);
}

extern "C" void kernel_launch(void* const* d_in, const int* in_sizes, int n_in,
                              void* d_out_, int out_size, void* d_ws, size_t ws_size,
                              hipStream_t stream)
{
  const float* x       = (const float*)d_in[0];
  const float* mem     = (const float*)d_in[1];
  const float* prevqk1 = (const float*)d_in[2];
  const float* prevqk2 = (const float*)d_in[3];
  const float* in_g    = (const float*)d_in[4];
  const float* in_b    = (const float*)d_in[5];
  const float* w_in    = (const float*)d_in[6];
  const float* mem_g   = (const float*)d_in[7];
  const float* mem_b   = (const float*)d_in[8];
  const float* w_mem   = (const float*)d_in[9];
  const float* a1_g    = (const float*)d_in[10];
  const float* a1_b    = (const float*)d_in[11];
  const float* w_a1    = (const float*)d_in[12];
  const float* b_a1    = (const float*)d_in[13];
  const float* wq1     = (const float*)d_in[14];
  const float* cq1     = (const float*)d_in[15];
  const float* wk1     = (const float*)d_in[16];
  const float* ck1     = (const float*)d_in[17];
  const float* wv1     = (const float*)d_in[18];
  const float* cv1     = (const float*)d_in[19];
  const float* wo1     = (const float*)d_in[20];
  const float* rel1    = (const float*)d_in[21];
  const float* a2_g    = (const float*)d_in[22];
  const float* a2_b    = (const float*)d_in[23];
  const float* w_a2    = (const float*)d_in[24];
  const float* b_a2    = (const float*)d_in[25];
  const float* wq2     = (const float*)d_in[26];
  const float* cq2     = (const float*)d_in[27];
  const float* wk2     = (const float*)d_in[28];
  const float* ck2     = (const float*)d_in[29];
  const float* wv2     = (const float*)d_in[30];
  const float* cv2     = (const float*)d_in[31];
  const float* wo2     = (const float*)d_in[32];
  const float* rel2    = (const float*)d_in[33];
  const float* n3_g    = (const float*)d_in[34];
  const float* n3_b    = (const float*)d_in[35];
  const float* w_ff    = (const float*)d_in[36];
  const float* b_ff    = (const float*)d_in[37];
  const float* w1      = (const float*)d_in[38];
  const float* w2      = (const float*)d_in[39];
  (void)in_sizes; (void)n_in; (void)out_size;

  float* out = (float*)d_out_;
  float* ws  = (float*)d_ws;

  // d_out layout: xc (B,5,BINS,CROP) | qk1 (B,NB,CROP,CROP) | qk2 (B,NB,CROP,CROP)
  const long XC   = (long)Bn * 5 * PL;
  const long QKN  = (long)Bn * NBn * PL;
  float* qk1_out  = out + XC;
  float* qk2_out  = out + XC + QKN;
  const long xcBS = 5 * PL;

  // ---------------- workspace carve-up (units: floats) ----------------
  long off = 0;
  float* mu_x   = ws + off; off += (long)Bn * CROPn;
  float* rs_x   = ws + off; off += (long)Bn * CROPn;
  float* mu_m   = ws + off; off += (long)Bn * CROPn;
  float* rs_m   = ws + off; off += (long)Bn * CROPn;
  float* mu_a1  = ws + off; off += (long)Bn * CROPn;
  float* rs_a1  = ws + off; off += (long)Bn * CROPn;
  float* mu_a2  = ws + off; off += (long)Bn * CROPn;
  float* rs_a2  = ws + off; off += (long)Bn * CROPn;
  float* mu_n3  = ws + off; off += (long)Bn * CROPn;
  float* rs_n3  = ws + off; off += (long)Bn * CROPn;
  off = 65536;
  // bf16 buffers (each Bn*PL bf16 = Bn*PL/2 float-units)
  __bf16* hinbf  = (__bf16*)(ws + off); off += Bn * PL / 2;
  __bf16* m1Tbf  = (__bf16*)(ws + off); off += Bn * PL / 2;
  __bf16* qb     = (__bf16*)(ws + off); off += Bn * PL / 2;
  __bf16* kb     = (__bf16*)(ws + off); off += Bn * PL / 2;
  __bf16* vb     = (__bf16*)(ws + off); off += Bn * PL / 2;
  __bf16* merged = (__bf16*)(ws + off); off += Bn * PL / 2;
  __bf16* probs  = (__bf16*)(ws + off); off += (long)Bn * NBn * PL / 2;  // also FFN bf16 act
  // bf16 weights, stored pre-transposed (N,K)
  __bf16* wqT1  = (__bf16*)(ws + off); off += PL / 2;
  __bf16* wkT1  = (__bf16*)(ws + off); off += PL / 2;
  __bf16* wvT1  = (__bf16*)(ws + off); off += PL / 2;
  __bf16* woT1  = (__bf16*)(ws + off); off += PL / 2;
  __bf16* wqT2  = (__bf16*)(ws + off); off += PL / 2;
  __bf16* wkT2  = (__bf16*)(ws + off); off += PL / 2;
  __bf16* wvT2  = (__bf16*)(ws + off); off += PL / 2;
  __bf16* woT2  = (__bf16*)(ws + off); off += PL / 2;
  __bf16* relT1 = (__bf16*)(ws + off); off += (long)HDn * CROPn / 2;  // (CROP, HD)
  __bf16* relT2 = (__bf16*)(ws + off); off += (long)HDn * CROPn / 2;
  __bf16* w1T   = (__bf16*)(ws + off); off += (long)BINSn * FFn / 2;  // (FF, BINS)
  __bf16* w2T   = (__bf16*)(ws + off); off += (long)BINSn * FFn / 2;  // (BINS, FF)
  // fp32 buffers
  float* tmp = ws + off; off += (long)Bn * PL;  // GEMM out pre-conv
  if ((size_t)off * sizeof(float) > ws_size) return;  // insufficient workspace

  dim3 sgrid(CROPn / 256, Bn), sblk(256);
  dim3 pgrid(CROPn / 256, BINSn, Bn);
  dim3 tgrid(CROPn / 32, BINSn / 32, Bn), tblk(32, 8);
  dim3 xgrid(32, 32, Bn);  // premix: (crop/32, bins/32, B)

  // ---- one-shot weight convert + transpose: W (R,C) fp32 -> W^T (C,R) bf16 ----
  auto tcvt = [&](const float* src, __bf16* dst, int R, int C) {
    transpose_bf_kernel<<<dim3(C / 32, R / 32, 1), tblk, 0, stream>>>(src, 0, dst, 0, R, C);
  };
  tcvt(wq1, wqT1, BINSn, BINSn); tcvt(wk1, wkT1, BINSn, BINSn);
  tcvt(wv1, wvT1, BINSn, BINSn); tcvt(wo1, woT1, BINSn, BINSn);
  tcvt(wq2, wqT2, BINSn, BINSn); tcvt(wk2, wkT2, BINSn, BINSn);
  tcvt(wv2, wvT2, BINSn, BINSn); tcvt(wo2, woT2, BINSn, BINSn);
  tcvt(rel1, relT1, HDn, CROPn); tcvt(rel2, relT2, HDn, CROPn);   // (HD,CROP)->(CROP,HD)
  tcvt(w1, w1T, BINSn, FFn);     tcvt(w2, w2T, FFn, BINSn);

  // ---- input norms + channel projections -> xc ch0/ch1 ----
  stats_kernel<<<sgrid, sblk, 0, stream>>>(x, (long)CHn * PL, PL, CHn, mu_x, rs_x);
  project_kernel<<<pgrid, sblk, 0, stream>>>(x, (long)CHn * PL, PL, CHn, mu_x, rs_x,
                                             in_g, in_b, w_in, out, xcBS);
  stats_kernel<<<sgrid, sblk, 0, stream>>>(mem, (long)CHn * PL, PL, CHn, mu_m, rs_m);
  project_kernel<<<pgrid, sblk, 0, stream>>>(mem, (long)CHn * PL, PL, CHn, mu_m, rs_m,
                                             mem_g, mem_b, w_mem, out + PL, xcBS);
  // m1T (bf16) = transpose of xc ch1: (b,bins,crop) -> (b,crop,bins)
  transpose_bf_kernel<<<tgrid, tblk, 0, stream>>>(out + PL, xcBS, m1Tbf, PL, BINSn, CROPn);

  // ---- self-attention branch ----
  stats_kernel<<<sgrid, sblk, 0, stream>>>(out, xcBS, PL, 2, mu_a1, rs_a1);
  premix_kernel<<<xgrid, tblk, 0, stream>>>(out, xcBS, PL, 2, mu_a1, rs_a1,
                                            a1_g, a1_b, w_a1, b_a1, hinbf);
  run_attention(stream, hinbf, hinbf, wqT1, cq1, wkT1, ck1, wvT1, cv1, woT1, relT1,
                prevqk1, qk1_out, out + 2 * PL, xcBS, tmp, qb, kb, vb, probs, merged);

  // ---- cross-attention branch ----
  stats_kernel<<<sgrid, sblk, 0, stream>>>(out, xcBS, PL, 3, mu_a2, rs_a2);
  premix_kernel<<<xgrid, tblk, 0, stream>>>(out, xcBS, PL, 3, mu_a2, rs_a2,
                                            a2_g, a2_b, w_a2, b_a2, hinbf);
  run_attention(stream, hinbf, m1Tbf, wqT2, cq2, wkT2, ck2, wvT2, cv2, woT2, relT2,
                prevqk2, qk2_out, out + 3 * PL, xcBS, tmp, qb, kb, vb, probs, merged);

  // ---- squared-ReLU FFN branch ----
  stats_kernel<<<sgrid, sblk, 0, stream>>>(out, xcBS, PL, 4, mu_n3, rs_n3);
  premix_kernel<<<xgrid, tblk, 0, stream>>>(out, xcBS, PL, 4, mu_n3, rs_n3,
                                            n3_g, n3_b, w_ff, b_ff, hinbf);
  __bf16* ffb2 = probs;  // bf16 hidden (b, CROP, FF): 8M of 16M bf16 available
  // hidden = relu(hin @ w1)^2, bf16, fused in epilogue
  launch_gemm(stream, true, CROPn, FFn, BINSn, hinbf, BINSn, PL, 0,
              w1T, BINSn, 0, 0, nullptr, ffb2, EP_BF16 | EP_RELU2, 1.0f, nullptr,
              FFn, (long)CROPn * FFn, 0, Bn, 1);
  // xc ch4 = (hidden @ w2)^T stored directly transposed
  launch_gemm(stream, true, CROPn, BINSn, FFn, ffb2, FFn, (long)CROPn * FFn, 0,
              w2T, FFn, 0, 0, out + 4 * PL, nullptr, EP_TRANSC, 1.0f, nullptr,
              CROPn, xcBS, 0, Bn, 1);
}
